// MSE_PQ_40243843563641
// MI455X (gfx1250) — matrix operand order repaired and checked by reading, not compiled, not run on hardware
//
#include <hip/hip_runtime.h>
#include <hip/hip_bf16.h>

typedef __attribute__((ext_vector_type(16))) __bf16 v16bf;
typedef __attribute__((ext_vector_type(8)))  __bf16 v8bf;
typedef __attribute__((ext_vector_type(8)))  float  v8f;

#define LEN_VEC 512
#define NQ      8
#define NCW     256
#define DSUB    64
#define CT_STRIDE 264   // padded k-stride (bf16 elems) for 32-row CT half
#define P_STRIDE  264   // padded k-stride (16-bit elems) for 16x256 P/dist tile

union ABReg {
    v16bf v;
    v8bf  h[2];
    __bf16 e[16];
};

__device__ __forceinline__ unsigned short f32_to_bf16_bits(float f) {
    __bf16 h = (__bf16)f;
    return __builtin_bit_cast(unsigned short, h);
}
__device__ __forceinline__ unsigned short f32_to_f16_bits(float f) {
    _Float16 h = (_Float16)f;
    return __builtin_bit_cast(unsigned short, h);
}
__device__ __forceinline__ float f16_bits_to_f32(unsigned short u) {
    _Float16 h = __builtin_bit_cast(_Float16, u);
    return (float)h;
}

__device__ __forceinline__ void cvt16(ABReg& dst, const float4* f) {
    #pragma unroll
    for (int j = 0; j < 4; ++j) {
        dst.e[4*j+0] = (__bf16)f[j].x;
        dst.e[4*j+1] = (__bf16)f[j].y;
        dst.e[4*j+2] = (__bf16)f[j].z;
        dst.e[4*j+3] = (__bf16)f[j].w;
    }
}

__global__ __launch_bounds__(128)
void pq_soft_wmma_kernel(const float* __restrict__ x,
                         const float* __restrict__ cb,
                         float* __restrict__ qx,
                         long long* __restrict__ ids,
                         const int* __restrict__ soft_rate_p,
                         int N)
{
    // LDS: 32x264 bf16 CT half (16896B) + 4 x 16x264 u16 P/dist (33792B) + c2 (1024B)
    __shared__ __align__(16) __bf16         sCT[32 * CT_STRIDE];
    __shared__ __align__(16) unsigned short sP[4][16 * P_STRIDE];
    __shared__ __align__(16) float          sC2[NCW];

    const int tid  = threadIdx.x;
    const int wave = tid >> 5;
    const int lane = tid & 31;
    const int ln   = lane & 15;   // column index within 16-wide tile
    const int g    = lane >> 4;   // lane half-group (wave32 => g in {0,1})
    const int n0   = blockIdx.x * 64 + wave * 16;

    unsigned short* myP = sP[wave];

    const float srate = (float)(*soft_rate_p);
    const bool  hard  = (srate < 0.0f);

    const v8f zero8 = {0.f,0.f,0.f,0.f,0.f,0.f,0.f,0.f};

    #pragma unroll 1
    for (int m = 0; m < NQ; ++m) {
        const float* cbm = cb + (size_t)m * NCW * DSUB;

        // ---------- cooperative: c2[k] and CT rows d=0..31 ----------
        for (int kk = tid; kk < NCW; kk += 128) {
            const float4* row = (const float4*)(cbm + (size_t)kk * DSUB);
            float acc2 = 0.f;
            #pragma unroll
            for (int j = 0; j < 16; ++j) {
                float4 v = row[j];
                acc2 += v.x*v.x + v.y*v.y + v.z*v.z + v.w*v.w;
                int d = j * 4;
                if (d < 32) {
                    sCT[(d+0)*CT_STRIDE + kk] = (__bf16)v.x;
                    sCT[(d+1)*CT_STRIDE + kk] = (__bf16)v.y;
                    sCT[(d+2)*CT_STRIDE + kk] = (__bf16)v.z;
                    sCT[(d+3)*CT_STRIDE + kk] = (__bf16)v.w;
                }
            }
            sC2[kk] = acc2;
        }
        __syncthreads();

        // ---------- A operands (x tile, 16-bit WMMA A layout) + fused x2 ----------
        const float* xrow = x + (size_t)(n0 + ln) * LEN_VEC + m * DSUB;
        ABReg A[2];
        float x2p = 0.f;
        #pragma unroll
        for (int s = 0; s < 2; ++s) {
            int d0 = s * 32 + g * 16;
            float4 f[4];
            f[0] = *(const float4*)(xrow + d0);
            f[1] = *(const float4*)(xrow + d0 + 4);
            f[2] = *(const float4*)(xrow + d0 + 8);
            f[3] = *(const float4*)(xrow + d0 + 12);
            #pragma unroll
            for (int j = 0; j < 4; ++j)
                x2p += f[j].x*f[j].x + f[j].y*f[j].y + f[j].z*f[j].z + f[j].w*f[j].w;
            cvt16(A[s], f);
        }
        // lane covers half of row ln's dims; xor-16 partner covers the other half
        float x2full = x2p + __shfl_xor(x2p, 16);
        float x2r[8];
        #pragma unroll
        for (int i = 0; i < 8; ++i) x2r[i] = __shfl(x2full, g * 8 + i);

        if (m + 1 < NQ) __builtin_prefetch(xrow + DSUB, 0, 1);

        // ---------- GEMM1 (pipelined): dist (fp16 -> LDS) + incremental argmin ----------
        float vmin[8]; int vidx[8];
        #pragma unroll
        for (int i = 0; i < 8; ++i) { vmin[i] = 3.4e38f; vidx[i] = 0; }

        float4 fb[8];   // rotating prefetch buffer: next tile's codebook rows (raw fp32)
        {
            const float* b0 = cbm + (size_t)ln * DSUB + g*16;
            #pragma unroll
            for (int j = 0; j < 4; ++j) fb[j]     = *(const float4*)(b0 + 4*j);
            #pragma unroll
            for (int j = 0; j < 4; ++j) fb[4 + j] = *(const float4*)(b0 + 32 + 4*j);
        }
        #pragma unroll
        for (int t = 0; t < 16; ++t) {
            ABReg B0, B1;
            cvt16(B0, &fb[0]);
            cvt16(B1, &fb[4]);
            if (t < 15) {   // prefetch tile t+1 (WAR on fb caps pipeline depth at 1)
                const float* b0 = cbm + (size_t)((t+1)*16 + ln) * DSUB + g*16;
                #pragma unroll
                for (int j = 0; j < 4; ++j) fb[j]     = *(const float4*)(b0 + 4*j);
                #pragma unroll
                for (int j = 0; j < 4; ++j) fb[4 + j] = *(const float4*)(b0 + 32 + 4*j);
            }
            v8f acc = zero8;
            acc = __builtin_amdgcn_wmma_f32_16x16x32_bf16(
                    false, A[0].v, false, B0.v, (short)0, acc, false, false);
            acc = __builtin_amdgcn_wmma_f32_16x16x32_bf16(
                    false, A[1].v, false, B1.v, (short)0, acc, false, false);
            float c2k = sC2[t*16 + ln];
            #pragma unroll
            for (int i = 0; i < 8; ++i) {
                float d2   = x2r[i] + c2k - 2.0f * acc[i];
                float dist = sqrtf(fmaxf(d2, 0.0f));
                if (dist < vmin[i]) { vmin[i] = dist; vidx[i] = t*16 + ln; }
                myP[(i + 8*g) * P_STRIDE + t*16 + ln] = f32_to_f16_bits(dist);
            }
        }

        // ---------- 16-lane butterfly min/argmin (wave32-safe, xor<=8) ----------
        #pragma unroll
        for (int msk = 1; msk <= 8; msk <<= 1) {
            #pragma unroll
            for (int i = 0; i < 8; ++i) {
                float ov = __shfl_xor(vmin[i], msk);
                int   oi = __shfl_xor(vidx[i], msk);
                if (ov < vmin[i] || (ov == vmin[i] && oi < vidx[i])) {
                    vmin[i] = ov; vidx[i] = oi;
                }
            }
        }
        #pragma unroll
        for (int i = 0; i < 8; ++i) {
            if (ln == i) ids[(size_t)m * N + n0 + g*8 + i] = (long long)vidx[i];
        }

        // ---------- softmax weights: fp16 dist -> bf16 P (same LDS slots) ----------
        float rsum[8];
        #pragma unroll
        for (int i = 0; i < 8; ++i) rsum[i] = 0.f;
        #pragma unroll
        for (int t = 0; t < 16; ++t) {
            #pragma unroll
            for (int i = 0; i < 8; ++i) {
                int off = (i + 8*g) * P_STRIDE + t*16 + ln;
                float dist = f16_bits_to_f32(myP[off]);
                float p;
                if (hard) p = ((t*16 + ln) == vidx[i]) ? 1.0f : 0.0f;
                else      p = __expf(-srate * (dist - vmin[i]));
                rsum[i] += p;
                myP[off] = f32_to_bf16_bits(p);
            }
        }
        #pragma unroll
        for (int msk = 1; msk <= 8; msk <<= 1) {
            #pragma unroll
            for (int i = 0; i < 8; ++i) rsum[i] += __shfl_xor(rsum[i], msk);
        }
        float rinv[8];
        #pragma unroll
        for (int i = 0; i < 8; ++i) rinv[i] = 1.0f / rsum[i];

        // ---------- GEMM2 (pipelined): Qx = P @ C, two d-halves (CT half-resident) ----------
        #pragma unroll
        for (int half = 0; half < 2; ++half) {
            if (half == 1) {
                __syncthreads();   // everyone done with CT rows 0..31
                for (int kk = tid; kk < NCW; kk += 128) {
                    const float4* row = (const float4*)(cbm + (size_t)kk * DSUB + 32);
                    #pragma unroll
                    for (int j = 0; j < 8; ++j) {
                        float4 v = row[j];
                        int d = j * 4;
                        sCT[(d+0)*CT_STRIDE + kk] = (__bf16)v.x;
                        sCT[(d+1)*CT_STRIDE + kk] = (__bf16)v.y;
                        sCT[(d+2)*CT_STRIDE + kk] = (__bf16)v.z;
                        sCT[(d+3)*CT_STRIDE + kk] = (__bf16)v.w;
                    }
                }
                __syncthreads();
            }
            // double-buffered LDS operands: loads for slab s+1 issue before WMMAs of slab s
            ABReg Ap[2], Bq0[2], Bq1[2];
            {
                const unsigned short* pb = myP + ln * P_STRIDE + g*16;
                Ap[0].h[0] = *(const v8bf*)pb;
                Ap[0].h[1] = *(const v8bf*)(pb + 8);
                const __bf16* c0p = sCT + ln * CT_STRIDE + g*16;
                Bq0[0].h[0] = *(const v8bf*)c0p;
                Bq0[0].h[1] = *(const v8bf*)(c0p + 8);
                const __bf16* c1p = sCT + (16 + ln) * CT_STRIDE + g*16;
                Bq1[0].h[0] = *(const v8bf*)c1p;
                Bq1[0].h[1] = *(const v8bf*)(c1p + 8);
            }
            v8f qa0 = zero8, qa1 = zero8;
            #pragma unroll
            for (int s = 0; s < 8; ++s) {       // 8 k-slabs of 32 over K=256
                const int cur = s & 1, nxt = cur ^ 1;
                if (s < 7) {
                    const unsigned short* pb = myP + ln * P_STRIDE + (s+1)*32 + g*16;
                    Ap[nxt].h[0] = *(const v8bf*)pb;
                    Ap[nxt].h[1] = *(const v8bf*)(pb + 8);
                    const __bf16* c0p = sCT + ln * CT_STRIDE + (s+1)*32 + g*16;
                    Bq0[nxt].h[0] = *(const v8bf*)c0p;
                    Bq0[nxt].h[1] = *(const v8bf*)(c0p + 8);
                    const __bf16* c1p = sCT + (16 + ln) * CT_STRIDE + (s+1)*32 + g*16;
                    Bq1[nxt].h[0] = *(const v8bf*)c1p;
                    Bq1[nxt].h[1] = *(const v8bf*)(c1p + 8);
                }
                qa0 = __builtin_amdgcn_wmma_f32_16x16x32_bf16(
                        false, Ap[cur].v, false, Bq0[cur].v, (short)0, qa0, false, false);
                qa1 = __builtin_amdgcn_wmma_f32_16x16x32_bf16(
                        false, Ap[cur].v, false, Bq1[cur].v, (short)0, qa1, false, false);
            }
            // normalize + store
            #pragma unroll
            for (int tt = 0; tt < 2; ++tt) {
                const v8f& qa = (tt == 0) ? qa0 : qa1;
                int dcol = half*32 + tt*16 + ln;
                #pragma unroll
                for (int i = 0; i < 8; ++i) {
                    int n = n0 + 8*g + i;
                    qx[(size_t)n * LEN_VEC + m * DSUB + dcol] = qa[i] * rinv[i];
                }
            }
        }
        __syncthreads();   // before next m overwrites CT / c2
    }
}

extern "C" void kernel_launch(void* const* d_in, const int* in_sizes, int n_in,
                              void* d_out, int out_size, void* d_ws, size_t ws_size,
                              hipStream_t stream) {
    const float* x    = (const float*)d_in[0];
    const float* cb   = (const float*)d_in[1];
    const int*   rate = (const int*)d_in[2];

    int N = in_sizes[0] / LEN_VEC;

    float* qx = (float*)d_out;
    long long* ids = (long long*)(qx + (size_t)N * LEN_VEC);

    int nblocks = N / 64;   // 64 vectors per block (4 waves x 16)
    pq_soft_wmma_kernel<<<nblocks, 128, 0, stream>>>(x, cb, qx, ids, rate, N);
}
